// NonLocalBlock_2233382994276
// MI455X (gfx1250) — matrix operand order repaired and checked by reading, not compiled, not run on hardware
//
#include <hip/hip_runtime.h>
#include <hip/hip_bf16.h>

typedef __attribute__((ext_vector_type(16))) _Float16 v16h;
typedef __attribute__((ext_vector_type(8)))  float    v8f;

#define B_    8
#define C_    512
#define CI_   256
#define N_    4096   // 64*64
#define M_    1024   // 32*32

// ---------------- WMMA fragment helpers (contiguous K order: k = 16*hi + s) ----------------

__device__ __forceinline__ v8f wmma_f16(v16h a, v16h b, v8f c) {
    return __builtin_amdgcn_wmma_f32_16x16x32_f16(
        /*neg_a=*/false, a, /*neg_b=*/false, b,
        /*c_mod=*/(short)0, c, /*reuse_a=*/false, /*reuse_b=*/false);
}

// A 16x32 f16: lane<16 -> row=lane, K = 0..15; lane>=16 -> row=lane-16, K = 16..31.
// rowPtr points at (row, k0) in a K-contiguous row. Pairs (2v,2v+1) are contiguous.
__device__ __forceinline__ v16h loadA16x32(const _Float16* rowPtr, int hi) {
    v16h a;
#pragma unroll
    for (int v = 0; v < 8; ++v) {
        int k = 2 * v + (hi << 4);
        a[2 * v]     = rowPtr[k];
        a[2 * v + 1] = rowPtr[k + 1];
    }
    return a;
}

// B 32x16 f16, K-contiguous storage: lane<16 col=lane K=0..15; lane>=16 K=16..31.
__device__ __forceinline__ v16h loadB_kcontig_h(const _Float16* colPtr, int hi) {
    const _Float16* p = colPtr + (hi ? 16 : 0);
    v16h b;
#pragma unroll
    for (int i = 0; i < 16; ++i) b[i] = p[i];
    return b;
}

// Same but source is f32 (weights), converted to f16 on the fly.
__device__ __forceinline__ v16h loadB_kcontig_f32(const float* colPtr, int hi) {
    const float* p = colPtr + (hi ? 16 : 0);
    v16h b;
#pragma unroll
    for (int i = 0; i < 16; ++i) b[i] = (_Float16)p[i];
    return b;
}

// CDNA5 async global->LDS copy (ASYNCcnt-tracked), GV addressing, 16B/lane.
__device__ __forceinline__ void async_lds_copy_b128(unsigned ldsOff, const void* gptr) {
    asm volatile("global_load_async_to_lds_b128 %0, %1, off"
                 :: "v"(ldsOff), "v"(gptr) : "memory");
}
__device__ __forceinline__ void wait_asynccnt0() {
    asm volatile("s_wait_asynccnt 0" ::: "memory");
}

// ---------------- Kernel 1: fused 1x1-conv projections (g / theta / phi) ---
// out[b][n][o] (f16) = sum_c x[b][c][n] * W[o][c] + bias[o]
// grid: (N/16, B, 3), block: 256 (8 waves); wave w owns o-tiles {2w, 2w+1}.
__global__ void proj_kernel(const float* __restrict__ x,
                            const float* __restrict__ wg,  const float* __restrict__ bg,
                            const float* __restrict__ wth, const float* __restrict__ bth,
                            const float* __restrict__ wph, const float* __restrict__ bph,
                            _Float16* __restrict__ g_full,
                            _Float16* __restrict__ th_full,
                            _Float16* __restrict__ ph_full)
{
    const int nTile = blockIdx.x;
    const int b     = blockIdx.y;
    const int proj  = blockIdx.z;

    const float* W;  const float* bias;  _Float16* out;
    if (proj == 0)      { W = wg;  bias = bg;  out = g_full;  }
    else if (proj == 1) { W = wth; bias = bth; out = th_full; }
    else                { W = wph; bias = bph; out = ph_full; }

    const int tid  = threadIdx.x;
    const int lane = tid & 31;
    const int wave = tid >> 5;
    const int hi   = lane >> 4;
    const int col  = lane & 15;
    const int row  = lane & 15;
    const int nBase = nTile * 16;
    const int o0    = wave * 32;

    __shared__ __align__(16) _Float16 As[16 * 32];   // 16 spatial rows x 32 channels

    v8f acc0 = {}, acc1 = {};
    const float* xb = x + (size_t)b * C_ * N_ + nBase;

    for (int k0 = 0; k0 < C_; k0 += 32) {
        // cooperative stage: 512 f32 -> f16, consecutive tids read contiguous n
        for (int i = tid; i < 512; i += 256) {
            int r = i & 15, k = i >> 4;
            As[r * 32 + k] = (_Float16)xb[(size_t)(k0 + k) * N_ + r];
        }
        __syncthreads();
        v16h a  = loadA16x32(As + row * 32, hi);
        v16h b0 = loadB_kcontig_f32(W + (size_t)(o0 + col) * C_ + k0, hi);
        acc0 = wmma_f16(a, b0, acc0);
        v16h b1 = loadB_kcontig_f32(W + (size_t)(o0 + 16 + col) * C_ + k0, hi);
        acc1 = wmma_f16(a, b1, acc1);
        __syncthreads();
    }

    _Float16* ob = out + ((size_t)b * N_ + nBase) * CI_;
    const float bv0 = bias[o0 + col];
    const float bv1 = bias[o0 + 16 + col];
#pragma unroll
    for (int r = 0; r < 8; ++r) {
        int rr = r + 8 * hi;
        ob[(size_t)rr * CI_ + o0 + col]      = (_Float16)(acc0[r] + bv0);
        ob[(size_t)rr * CI_ + o0 + 16 + col] = (_Float16)(acc1[r] + bv1);
    }
}

// ---------------- Kernel 2a: 2x2 max-pool, [B][N][CI] -> [B][M][CI] --------
__global__ void pool_kernel(const _Float16* __restrict__ full,
                            _Float16* __restrict__ pooled)
{
    int idx = blockIdx.x * blockDim.x + threadIdx.x;   // over B*M*CI
    int ci = idx & (CI_ - 1);
    int m  = (idx >> 8) & (M_ - 1);
    int b  = idx >> 18;
    int ph = m >> 5, pw = m & 31;
    int n00 = (ph * 2) * 64 + pw * 2;
    const _Float16* fb = full + (size_t)b * N_ * CI_ + ci;
    float v0 = (float)fb[(size_t)n00 * CI_];
    float v1 = (float)fb[(size_t)(n00 + 1) * CI_];
    float v2 = (float)fb[(size_t)(n00 + 64) * CI_];
    float v3 = (float)fb[(size_t)(n00 + 65) * CI_];
    pooled[idx] = (_Float16)fmaxf(fmaxf(v0, v1), fmaxf(v2, v3));
}

// ---------------- Kernel 2b: 2x2 max-pool transposed, -> [B][CI][M] --------
// m-contiguous layout so the attention's second GEMM B-operand (K = m) is
// K-contiguous -> 32B/lane vector loads instead of 16 strided u16 loads.
__global__ void pool_kernel_t(const _Float16* __restrict__ full,
                              _Float16* __restrict__ pooledT)
{
    int idx = blockIdx.x * blockDim.x + threadIdx.x;   // over B*CI*M
    int m  = idx & (M_ - 1);
    int ci = (idx >> 10) & (CI_ - 1);
    int b  = idx >> 18;
    int ph = m >> 5, pw = m & 31;
    int n00 = (ph * 2) * 64 + pw * 2;
    const _Float16* fb = full + (size_t)b * N_ * CI_ + ci;
    float v0 = (float)fb[(size_t)n00 * CI_];
    float v1 = (float)fb[(size_t)(n00 + 1) * CI_];
    float v2 = (float)fb[(size_t)(n00 + 64) * CI_];
    float v3 = (float)fb[(size_t)(n00 + 65) * CI_];
    pooledT[idx] = (_Float16)fmaxf(fmaxf(v0, v1), fmaxf(v2, v3));
}

// ---------------- Kernel 3: flash-style attention per 16-row n-tile --------
// S[n,m] = sum_ci th[n,ci]*phi[m,ci];  P = softmax_m(S);  y[n,ci] = P @ g
// grid: (N/16, B), block: 128 (4 waves).
__global__ void attn_kernel(const _Float16* __restrict__ th,    // [B][N][CI]
                            const _Float16* __restrict__ phiP,  // [B][M][CI]
                            const _Float16* __restrict__ gT,    // [B][CI][M]
                            _Float16* __restrict__ y)           // [B][N][CI]
{
    const int nTile = blockIdx.x;
    const int b     = blockIdx.y;
    const int tid   = threadIdx.x;
    const int lane  = tid & 31;
    const int wave  = tid >> 5;
    const int hi    = lane >> 4;
    const int col   = lane & 15;
    const int row   = lane & 15;
    const int nBase = nTile * 16;

    __shared__ __align__(16) _Float16 Ath[16 * CI_];   // theta tile (8 KB)
    __shared__ float    wMax[4][16], wSum[4][16];
    __shared__ float    rowMax[16], rowInv[16];
    __shared__ __align__(16) _Float16 Pbuf[4][16 * 32]; // normalized P chunks (4 KB)

    // stage theta tile via CDNA5 async global->LDS DMA (16 B per lane per issue)
    const _Float16* thb = th + ((size_t)b * N_ + nBase) * CI_;
    for (int i = tid; i < (16 * CI_) / 8; i += 128) {
        unsigned ldsOff = (unsigned)(uintptr_t)&Ath[i * 8];
        async_lds_copy_b128(ldsOff, (const void*)(thb + i * 8));
    }
    wait_asynccnt0();
    __syncthreads();

    const _Float16* phb = phiP + (size_t)b * M_ * CI_;
    const _Float16* gbt = gT   + (size_t)b * CI_ * M_;

    // ---- pass 1: row statistics ----
    float runM[8], runS[8];
#pragma unroll
    for (int r = 0; r < 8; ++r) { runM[r] = -1e30f; runS[r] = 0.0f; }

    for (int mt = wave; mt < 64; mt += 4) {
        const int mBase = mt * 16;
        v8f s = {};
#pragma unroll
        for (int kk = 0; kk < 8; ++kk) {
            const int k0 = kk * 32;
            v16h a  = loadA16x32(Ath + row * CI_ + k0, hi);
            v16h bb = loadB_kcontig_h(phb + (size_t)(mBase + col) * CI_ + k0, hi);
            s = wmma_f16(a, bb, s);
        }
#pragma unroll
        for (int r = 0; r < 8; ++r) {
            float v = s[r];
#pragma unroll
            for (int off = 1; off < 16; off <<= 1) v = fmaxf(v, __shfl_xor(v, off));
            const float newM = fmaxf(runM[r], v);
            float e = __expf(s[r] - newM);
#pragma unroll
            for (int off = 1; off < 16; off <<= 1) e += __shfl_xor(e, off);
            runS[r] = runS[r] * __expf(runM[r] - newM) + e;
            runM[r] = newM;
        }
    }
    if ((lane & 15) == 0) {
#pragma unroll
        for (int r = 0; r < 8; ++r) {
            wMax[wave][r + 8 * hi] = runM[r];
            wSum[wave][r + 8 * hi] = runS[r];
        }
    }
    __syncthreads();
    if (tid < 16) {
        float Mx = -1e30f;
        for (int w = 0; w < 4; ++w) Mx = fmaxf(Mx, wMax[w][tid]);
        float S = 0.0f;
        for (int w = 0; w < 4; ++w) S += wSum[w][tid] * __expf(wMax[w][tid] - Mx);
        rowMax[tid] = Mx;
        rowInv[tid] = 1.0f / S;
    }
    __syncthreads();

    // ---- pass 2: y = P @ g ----
    v8f yacc[4] = {};
    const int ciBase = wave * 64;            // 4 ci-tiles per wave

    for (int grp = 0; grp < 8; ++grp) {
        const int mChunk = (grp * 4 + wave) * 32;
#pragma unroll
        for (int half = 0; half < 2; ++half) {
            const int mBase = mChunk + half * 16;
            v8f s = {};
#pragma unroll
            for (int kk = 0; kk < 8; ++kk) {
                const int k0 = kk * 32;
                v16h a  = loadA16x32(Ath + row * CI_ + k0, hi);
                v16h bb = loadB_kcontig_h(phb + (size_t)(mBase + col) * CI_ + k0, hi);
                s = wmma_f16(a, bb, s);
            }
#pragma unroll
            for (int r = 0; r < 8; ++r) {
                const int rr = r + 8 * hi;
                const float p = __expf(s[r] - rowMax[rr]) * rowInv[rr];
                Pbuf[wave][rr * 32 + half * 16 + col] = (_Float16)p;
            }
        }
        __syncthreads();
        for (int c = 0; c < 4; ++c) {
            const int mB = (grp * 4 + c) * 32;
            v16h a = loadA16x32(Pbuf[c] + row * 32, hi);
#pragma unroll
            for (int t = 0; t < 4; ++t) {
                const int ci = ciBase + t * 16 + col;
                v16h bb = loadB_kcontig_h(gbt + (size_t)ci * M_ + mB, hi);
                yacc[t] = wmma_f16(a, bb, yacc[t]);
            }
        }
        __syncthreads();
    }

    _Float16* yb = y + ((size_t)b * N_ + nBase) * CI_;
#pragma unroll
    for (int t = 0; t < 4; ++t)
#pragma unroll
        for (int r = 0; r < 8; ++r) {
            const int rr = r + 8 * hi;
            yb[(size_t)rr * CI_ + ciBase + t * 16 + col] = (_Float16)yacc[t][r];
        }
}

// ---------------- Kernel 4: output conv + folded BN + residual -------------
// out[b][co][n] = acc(n,co)*aScale[co] + bShift[co] + x[b][co][n]
// grid: (N/16, B), block: 256 (8 waves, 4 co-tiles each).
__global__ void outconv_kernel(const _Float16* __restrict__ y,   // [B][N][CI]
                               const float* __restrict__ ww,     // [C][CI]
                               const float* __restrict__ wb,
                               const float* __restrict__ gamma,
                               const float* __restrict__ beta,
                               const float* __restrict__ mean,
                               const float* __restrict__ var,
                               const float* __restrict__ x,
                               float* __restrict__ out)
{
    const int nTile = blockIdx.x;
    const int b     = blockIdx.y;
    const int tid   = threadIdx.x;
    const int lane  = tid & 31;
    const int wave  = tid >> 5;
    const int hi    = lane >> 4;
    const int col   = lane & 15;
    const int row   = lane & 15;
    const int nBase  = nTile * 16;
    const int coBase = wave * 64;

    const _Float16* yb = y + ((size_t)b * N_ + nBase) * CI_;

    v8f acc[4] = {};
    for (int k0 = 0; k0 < CI_; k0 += 32) {
        v16h a = loadA16x32(yb + row * CI_ + k0, hi);
#pragma unroll
        for (int t = 0; t < 4; ++t) {
            v16h bb = loadB_kcontig_f32(ww + (size_t)(coBase + t * 16 + col) * CI_ + k0, hi);
            acc[t] = wmma_f16(a, bb, acc[t]);
        }
    }
#pragma unroll
    for (int t = 0; t < 4; ++t) {
        const int co = coBase + t * 16 + col;
        const float inv = rsqrtf(var[co] + 1e-5f);
        const float aS  = gamma[co] * inv;
        const float bS  = beta[co] + (wb[co] - mean[co]) * aS;
#pragma unroll
        for (int r = 0; r < 8; ++r) {
            const int rr = r + 8 * hi;
            const size_t oi = ((size_t)(b * C_ + co)) * N_ + nBase + rr;
            out[oi] = acc[t][r] * aS + bS + x[oi];
        }
    }
}

// ---------------- host launcher --------------------------------------------
extern "C" void kernel_launch(void* const* d_in, const int* in_sizes, int n_in,
                              void* d_out, int out_size, void* d_ws, size_t ws_size,
                              hipStream_t stream) {
    const float* x    = (const float*)d_in[0];
    const float* g_w  = (const float*)d_in[1];
    const float* g_b  = (const float*)d_in[2];
    const float* th_w = (const float*)d_in[3];
    const float* th_b = (const float*)d_in[4];
    const float* ph_w = (const float*)d_in[5];
    const float* ph_b = (const float*)d_in[6];
    const float* w_w  = (const float*)d_in[7];
    const float* w_b  = (const float*)d_in[8];
    const float* bn_g = (const float*)d_in[9];
    const float* bn_b = (const float*)d_in[10];
    const float* bn_m = (const float*)d_in[11];
    const float* bn_v = (const float*)d_in[12];
    float* out = (float*)d_out;

    const size_t fullBytes   = (size_t)B_ * N_ * CI_ * sizeof(_Float16);  // 16.78 MB
    const size_t pooledBytes = (size_t)B_ * M_ * CI_ * sizeof(_Float16);  //  4.19 MB

    char* ws = (char*)d_ws;
    _Float16* g_full  = (_Float16*)ws;  ws += fullBytes;
    _Float16* th_full = (_Float16*)ws;  ws += fullBytes;
    _Float16* ph_full = (_Float16*)ws;  ws += fullBytes;
    _Float16* phiP    = (_Float16*)ws;  ws += pooledBytes;   // [B][M][CI]
    _Float16* gTp     = (_Float16*)ws;  ws += pooledBytes;   // [B][CI][M]
    _Float16* yh      = (_Float16*)ws;  ws += fullBytes;

    proj_kernel<<<dim3(N_ / 16, B_, 3), 256, 0, stream>>>(
        x, g_w, g_b, th_w, th_b, ph_w, ph_b, g_full, th_full, ph_full);

    const int poolElems = B_ * M_ * CI_;
    pool_kernel<<<poolElems / 256, 256, 0, stream>>>(ph_full, phiP);
    pool_kernel_t<<<poolElems / 256, 256, 0, stream>>>(g_full, gTp);

    attn_kernel<<<dim3(N_ / 16, B_), 128, 0, stream>>>(th_full, phiP, gTp, yh);

    outconv_kernel<<<dim3(N_ / 16, B_), 256, 0, stream>>>(
        yh, w_w, w_b, bn_g, bn_b, bn_m, bn_v, x, out);
}
